// MPS_57071525429423
// MI455X (gfx1250) — matrix operand order repaired and checked by reading, not compiled, not run on hardware
//
#include <hip/hip_runtime.h>

// MPS chain classifier on MI455X (gfx1250).
// B=16384 batch rows, N=784 sites, P=2, D=32, C=10.
// Per middle site per 16-row batch tile:
//   u (16x64, bf16) @ A_mid[n] (64x32, bf16) -> new_left (16x32, f32)
// = 4x v_wmma_f32_16x16x32_bf16 (2 K-chunks x 2 N-tiles), f32 accumulation.

typedef __attribute__((ext_vector_type(16))) __bf16 v16bf;
typedef __attribute__((ext_vector_type(8)))  float  v8f;
typedef __attribute__((ext_vector_type(4)))  float  v4f;

#define NSITES 784
#define NMID   782      // sites 1..782
#define DBOND  32

// ---------------------------------------------------------------------------
// Prepass: convert A_mid (f32, [782][32][2][32]) into bf16 WMMA B-fragments.
// Fragment layout: [n][kc][nt][lane][slot]  (slot = 16 bf16 = one v16bf/lane)
// B k order chosen here (contiguous, not ISA table; A matches): lanes 0-15 hold K'=0..15,
// lanes 16-31 hold K'=16..31, N = lane%16, slot s -> K' = 16*(lane/16) + s.
// Matrix element: Amat[k][r] with k = 32*kc + K', r = 16*nt + lane%16,
// Amat[k][r] = A_mid[n][k>>1][k&1][r]   (flatten (l,p) -> k = 2l+p).
// ---------------------------------------------------------------------------
__global__ __launch_bounds__(256) void mps_pack_amid(const float* __restrict__ A_mid,
                                                     __bf16* __restrict__ Bfrag) {
    int tid = blockIdx.x * 256 + threadIdx.x;
    const int total = NMID * 2048;           // 782*4*32*16
    if (tid >= total) return;
    int s    = tid & 15;
    int lane = (tid >> 4) & 31;
    int nt   = (tid >> 9) & 1;
    int kc   = (tid >> 10) & 1;
    int n    = tid >> 11;
    int Kp   = 16 * (lane >> 4) + s;
    int k    = 32 * kc + Kp;
    int r    = 16 * nt + (lane & 15);
    float v  = A_mid[(size_t)n * 2048 + (size_t)(k >> 1) * 64 + (size_t)(k & 1) * 32 + r];
    Bfrag[tid] = (__bf16)v;
}

// ---------------------------------------------------------------------------
// Main chain kernel. 256 threads = 8 wave32; each wave owns 16 batch rows.
// leftT[l][m] (32x16 f32) lives in a private LDS slice per wave.
// Same-wave LDS RAW across sites relies on in-order DS execution (no
// explicit dscnt drain needed); wave_barrier() pins compiler scheduling.
// ---------------------------------------------------------------------------
__global__ __launch_bounds__(256) void mps_chain(const float* __restrict__ x,
                                                 const float* __restrict__ A_first,
                                                 const __bf16* __restrict__ Bfrag,
                                                 const float* __restrict__ A_last,
                                                 float* __restrict__ out) {
    __shared__ float ldsLeft[8][512];   // [wave][l*16 + m]
    __shared__ float xs0[8][32];        // [wave][m*2 + p]  site-0 x staging

    const int tid  = threadIdx.x;
    const int wave = tid >> 5;
    const int lane = tid & 31;
    const int m    = lane & 15;         // batch row within tile / N column
    const int g    = lane >> 4;         // lane half
    const int b0   = (blockIdx.x * 8 + wave) * 16;

    float* L = ldsLeft[wave];
    const float* xr = x + (size_t)(b0 + m) * (NSITES * 2);

    // ---- site 0: left[b,r] = x[b,0,0]*A_first[0,r] + x[b,0,1]*A_first[1,r]
    if (g == 0) { xs0[wave][m * 2 + 0] = xr[0]; xs0[wave][m * 2 + 1] = xr[1]; }
    __builtin_amdgcn_wave_barrier();
    asm volatile("s_wait_dscnt 0x0" ::: "memory");
    {
        float a0 = A_first[lane];          // A_first[0][l], l = lane
        float a1 = A_first[32 + lane];     // A_first[1][l]
        #pragma unroll
        for (int mm = 0; mm < 16; ++mm)
            L[lane * 16 + mm] = xs0[wave][mm * 2] * a0 + xs0[wave][mm * 2 + 1] * a1;
    }
    __builtin_amdgcn_wave_barrier();
    asm volatile("s_wait_dscnt 0x0" ::: "memory");

    // ---- middle sites: loop index n = 0..781 maps to site n+1 ----
    const float* xsite = xr + 2;  // x[b, 1, :]

    auto site_body = [&](int n, float x0, float x1) {
        // B fragments (bf16, pre-swizzled, L2-resident): 32B contiguous/lane
        const __bf16* fb = Bfrag + (size_t)n * 2048 + (size_t)lane * 16;
        v16bf Bf00 = *(const v16bf*)(fb);                 // kc=0, nt=0
        v16bf Bf01 = *(const v16bf*)(fb + 512);           // kc=0, nt=1
        v16bf Bf10 = *(const v16bf*)(fb + 1024);          // kc=1, nt=0
        v16bf Bf11 = *(const v16bf*)(fb + 1536);          // kc=1, nt=1

        // A-operand u (16x64 -> two 16x32 bf16 chunks), gathered from leftT.
        // A k order chosen here (matches B packing): M = lane%16; slot s -> K' = 16*g + s.
        v16bf uA0, uA1;
        #pragma unroll
        for (int s = 0; s < 16; ++s) {
            int Kp = 16 * g + s;
            {   int k = Kp;                               // chunk 0
                float lv = L[(k >> 1) * 16 + m];
                uA0[s] = (__bf16)(lv * ((k & 1) ? x1 : x0));
            }
            {   int k = 32 + Kp;                          // chunk 1
                float lv = L[(k >> 1) * 16 + m];
                uA1[s] = (__bf16)(lv * ((k & 1) ? x1 : x0));
            }
        }

        v8f c0 = {}, c1 = {};
        c0 = __builtin_amdgcn_wmma_f32_16x16x32_bf16(false, uA0, false, Bf00, (short)0, c0, false, false);
        c0 = __builtin_amdgcn_wmma_f32_16x16x32_bf16(false, uA1, false, Bf10, (short)0, c0, false, false);
        c1 = __builtin_amdgcn_wmma_f32_16x16x32_bf16(false, uA0, false, Bf01, (short)0, c1, false, false);
        c1 = __builtin_amdgcn_wmma_f32_16x16x32_bf16(false, uA1, false, Bf11, (short)0, c1, false, false);

        // Store new leftT: D layout (M,N): lane = N + 16*(M>=8), vgpr j = M%8.
        // Tile nt covers l = 16*nt + lane%16; per lane 8 contiguous floats.
        __builtin_amdgcn_wave_barrier();
        {
            float* p0 = &L[m * 16 + 8 * g];               // l = m (tile 0)
            *(v4f*)(p0)     = (v4f){c0[0], c0[1], c0[2], c0[3]};
            *(v4f*)(p0 + 4) = (v4f){c0[4], c0[5], c0[6], c0[7]};
            float* p1 = &L[(16 + m) * 16 + 8 * g];        // l = 16+m (tile 1)
            *(v4f*)(p1)     = (v4f){c1[0], c1[1], c1[2], c1[3]};
            *(v4f*)(p1 + 4) = (v4f){c1[4], c1[5], c1[6], c1[7]};
        }
        __builtin_amdgcn_wave_barrier();   // in-order DS gives RAW visibility
    };

    // WGP-scope prefetch (scope field 0 = default): pulls into all cache
    // levels; lane*128 spreads 32 lanes across the full 4KB site block.
    auto prefetch_site = [&](int pfn) {
        const char* p = (const char*)Bfrag + (size_t)pfn * 4096 + (size_t)lane * 128;
        asm volatile("global_prefetch_b8 %0, off" :: "v"(p));
    };

    // Peel n=0 (x at 8B-misaligned offset) so paired loads are 16B aligned.
    {
        float2 xp = *(const float2*)(xsite);
        site_body(0, xp.x, xp.y);
    }
    // Pairs n = 1,3,...,779  (x bytes at 16B-aligned offsets)
    for (int n = 1; n < NMID - 1; n += 2) {
        int pf = (n + 6 < NMID - 1) ? (n + 6) : (NMID - 2);
        prefetch_site(pf);
        prefetch_site(pf + 1);
        v4f xq = *(const v4f*)(xsite + 2 * n);            // sites n+1, n+2
        site_body(n,     xq[0], xq[1]);
        site_body(n + 1, xq[2], xq[3]);
    }
    // Peel n = 781 (last middle site)
    {
        float2 xp = *(const float2*)(xsite + 2 * (NMID - 1));
        site_body(NMID - 1, xp.x, xp.y);
    }

    // ---- last site: out[b,c] = sum_{l,p} left[b,l]*A_last[l,p,c]*x[b,783,p]
    asm volatile("s_wait_dscnt 0x0" ::: "memory");
    if (g == 0) {
        float x0 = xr[783 * 2 + 0];
        float x1 = xr[783 * 2 + 1];
        float acc[10];
        #pragma unroll
        for (int c = 0; c < 10; ++c) acc[c] = 0.f;
        for (int l = 0; l < DBOND; ++l) {
            float fl = L[l * 16 + m];
            const float* Al = A_last + l * 20;            // A_last[l][p][c]
            #pragma unroll
            for (int c = 0; c < 10; ++c)
                acc[c] += fl * (x0 * Al[c] + x1 * Al[10 + c]);
        }
        float* op = out + (size_t)(b0 + m) * 10;
        #pragma unroll
        for (int c = 0; c < 10; ++c) op[c] = acc[c];
    }
}

// ---------------------------------------------------------------------------
extern "C" void kernel_launch(void* const* d_in, const int* in_sizes, int n_in,
                              void* d_out, int out_size, void* d_ws, size_t ws_size,
                              hipStream_t stream) {
    const float* x       = (const float*)d_in[0];   // (16384, 784, 2)
    const float* A_first = (const float*)d_in[1];   // (2, 32)
    const float* A_mid   = (const float*)d_in[2];   // (782, 32, 2, 32)
    const float* A_last  = (const float*)d_in[3];   // (32, 2, 10)
    __bf16* Bfrag = (__bf16*)d_ws;                  // 782*2048*2 B = 3.2 MB

    const int total = NMID * 2048;
    mps_pack_amid<<<(total + 255) / 256, 256, 0, stream>>>(A_mid, Bfrag);
    mps_chain<<<128, 256, 0, stream>>>(x, A_first, Bfrag, A_last, (float*)d_out);
}
